// _Memory_Block_28879360098415
// MI455X (gfx1250) — matrix operand order repaired and checked by reading, not compiled, not run on hardware
//
#include <hip/hip_runtime.h>
#include <hip/hip_bf16.h>
#include <stdint.h>

typedef __bf16 bf16_t;
typedef __attribute__((ext_vector_type(16))) __bf16 v16bf;
typedef __attribute__((ext_vector_type(8)))  __bf16 v8bf;
typedef __attribute__((ext_vector_type(8)))  float  v8f;
typedef __attribute__((ext_vector_type(4))) unsigned int u32x4;
typedef __attribute__((ext_vector_type(8))) int          i32x8;
typedef __attribute__((ext_vector_type(4))) int          i32x4;

#define NN 65536
#define CC 512
#define KK 1024
#define RATE_F 0.999f

// ---------------------------------------------------------------- fragments
static __device__ __forceinline__ v16bf frag_from_chunks(v8bf c0, v8bf c1) {
  v16bf f;
#pragma unroll
  for (int i = 0; i < 8; ++i) { f[i] = c0[i]; f[i + 8] = c1[i]; }
  return f;
}

// A fragment 16x32 bf16, row-major source (ld elements per row).
// lanes 0-15 row=l hold K {0..7,16..23}; lanes 16-31 hold K {8..15,24..31}.
static __device__ __forceinline__ v16bf load_a_frag(const bf16_t* base, int ld,
                                                    int row0, int k0, int lane) {
  int half = lane >> 4, lr = lane & 15;
  const bf16_t* p = base + (size_t)(row0 + lr) * ld + k0 + half * 8;
  return frag_from_chunks(*(const v8bf*)p, *(const v8bf*)(p + 16));
}

// B fragment 32x16 bf16 from LDS tile: column n = lane%16 <- tile row lane%16,
// lanes 0-15 hold K {0..7,16..23}; lanes 16-31 hold K {8..15,24..31}.
static __device__ __forceinline__ v16bf load_b_frag_lds(const bf16_t* lds, int ld,
                                                        int k0, int lane) {
  int half = lane >> 4, lr = lane & 15;
  const bf16_t* p = lds + lr * ld + k0 + half * 8;
  return frag_from_chunks(*(const v8bf*)p, *(const v8bf*)(p + 16));
}

// ---------------------------------------------------------------- TDM (async tensor -> LDS)
// 2D tile load, data_size = 4B units. If pad=true, 16B LDS padding per 1024B row
// (pad_interval code 7 = 256 DWORDs, pad_amount code 3 = 4 DWORDs).
static __device__ __forceinline__ void tdm_load_2d(const void* gsrc, void* lds_dst,
                                                   unsigned width_u, unsigned rows,
                                                   unsigned stride_u, bool pad) {
  uint64_t ga = (uint64_t)(uintptr_t)gsrc;
  unsigned lo = (unsigned)(uintptr_t)lds_dst;           // flat low 32 = LDS offset
  u32x4 g0;
  g0[0] = 1u;                                           // count=1 (valid user D#)
  g0[1] = lo;                                           // lds_addr
  g0[2] = (unsigned)ga;                                 // global_addr[31:0]
  g0[3] = (unsigned)((ga >> 32) & 0x01FFFFFFu) | 0x80000000u; // addr[56:32] | type=2
  unsigned w0 = (2u << 16);                             // data_size = 4B
  if (pad) w0 |= (1u << 20) | (7u << 22) | (3u << 25);  // pad_en, interval, amount
  i32x8 g1;
  g1[0] = (int)w0;
  g1[1] = (int)((width_u & 0xFFFFu) << 16);                               // tensor_dim0 lo
  g1[2] = (int)(((width_u >> 16) & 0xFFFFu) | ((rows & 0xFFFFu) << 16));  // dim0 hi | dim1 lo
  g1[3] = (int)(((rows >> 16) & 0xFFFFu) | ((width_u & 0xFFFFu) << 16));  // dim1 hi | tile_dim0
  g1[4] = (int)(rows & 0xFFFFu);                                          // tile_dim1 (tile_dim2=0)
  g1[5] = (int)stride_u;                                                  // tensor_dim0_stride lo
  g1[6] = 0;                                                              // stride hi | dim1_stride lo
  g1[7] = 0;
  i32x4 z4 = {0, 0, 0, 0};
#if __clang_major__ >= 23
  i32x8 z8 = {0, 0, 0, 0, 0, 0, 0, 0};
  __builtin_amdgcn_tensor_load_to_lds(g0, g1, z4, z4, z8, 0);
#else
  __builtin_amdgcn_tensor_load_to_lds(g0, g1, z4, z4, 0);
#endif
}

// ---------------------------------------------------------------- utility
__global__ void zero_f32(float* p, int n) {
  int i = blockIdx.x * blockDim.x + threadIdx.x;
  if (i < n) p[i] = 0.0f;
}

// One wave per row: l2-normalize (nrows x 512) f32 into bf16; optional raw norms.
__global__ void row_normalize512(const float* __restrict__ src, bf16_t* __restrict__ dst,
                                 float* __restrict__ norms, int nrows) {
  int row = blockIdx.x * (blockDim.x >> 5) + (threadIdx.x >> 5);
  int lane = threadIdx.x & 31;
  if (row >= nrows) return;
  const float* p = src + (size_t)row * CC;
  float ss = 0.0f;
#pragma unroll
  for (int i = lane; i < CC; i += 32) { float v = p[i]; ss += v * v; }
#pragma unroll
  for (int off = 16; off; off >>= 1) ss += __shfl_xor(ss, off, 32);
  float nrm = sqrtf(ss);
  if (norms && lane == 0) norms[row] = nrm;
  float inv = 1.0f / fmaxf(nrm, 1e-12f);
  bf16_t* q = dst + (size_t)row * CC;
#pragma unroll
  for (int i = lane; i < CC; i += 32) q[i] = (bf16_t)(p[i] * inv);
}

// ---------------------------------------------------------------- GEMM1 + argmax
// Block = 8 waves x 16-row strips (128 rows). B tiles (16x512 bf16) TDM-staged,
// double-buffered in LDS, shared by all 8 waves. B fragments software-pipelined
// in registers so ds_load of frag k+1 overlaps the WMMA on frag k.
#define BSTR (CC + 8)   // padded LDS row stride (elements): 1024B + 16B TDM pad
__global__ __launch_bounds__(256) void gemm_score_argmax(const bf16_t* __restrict__ xn,
                                                         const bf16_t* __restrict__ mn,
                                                         int* __restrict__ embed_ind) {
  __shared__ __align__(16) bf16_t btile[2][16 * BSTR];
  int tid = threadIdx.x, wave = tid >> 5, lane = tid & 31;
  int row0 = blockIdx.x * 128 + wave * 16;
  bool leader = (__builtin_amdgcn_readfirstlane((int)threadIdx.x) == 0);

  if (leader) tdm_load_2d(mn, &btile[0][0], CC / 2, 16, CC / 2, true);

  v16bf afrag[CC / 32];
#pragma unroll
  for (int ks = 0; ks < CC / 32; ++ks) afrag[ks] = load_a_frag(xn, CC, row0, ks * 32, lane);

  float bestVal[8]; int bestIdx[8];
#pragma unroll
  for (int r = 0; r < 8; ++r) { bestVal[r] = -3.4e38f; bestIdx[r] = 0; }

  for (int jt = 0; jt < KK / 16; ++jt) {
    int cur = jt & 1;
    if (leader) {
      if (jt + 1 < KK / 16) {
        tdm_load_2d(mn + (size_t)(jt + 1) * 16 * CC, &btile[cur ^ 1][0], CC / 2, 16, CC / 2, true);
        __builtin_amdgcn_s_wait_tensorcnt(1);
      } else {
        __builtin_amdgcn_s_wait_tensorcnt(0);
      }
    }
    __syncthreads();
    const bf16_t* bb = &btile[cur][0];
    v8f acc = {};
    v16bf bcur = load_b_frag_lds(bb, BSTR, 0, lane);
#pragma unroll
    for (int ks = 0; ks < CC / 32; ++ks) {
      v16bf bnext = bcur;
      if (ks + 1 < CC / 32) bnext = load_b_frag_lds(bb, BSTR, (ks + 1) * 32, lane);
      acc = __builtin_amdgcn_wmma_f32_16x16x32_bf16(false, afrag[ks], false, bcur,
                                                    (short)0, acc, false, false);
      bcur = bnext;
    }
#pragma unroll
    for (int r = 0; r < 8; ++r) {
      float v = acc[r];
      int idx = jt * 16 + (lane & 15);
#pragma unroll
      for (int off = 8; off; off >>= 1) {
        float ov = __shfl_xor(v, off, 32);
        int   oi = __shfl_xor(idx, off, 32);
        if (ov > v || (ov == v && oi < idx)) { v = ov; idx = oi; }
      }
      if (v > bestVal[r] || (v == bestVal[r] && idx < bestIdx[r])) {
        bestVal[r] = v; bestIdx[r] = idx;
      }
    }
    __syncthreads();
  }
  if ((lane & 15) == 0) {
    int base = row0 + (lane >> 4) * 8;
#pragma unroll
    for (int r = 0; r < 8; ++r) embed_ind[base + r] = bestIdx[r];
  }
}

// ---------------------------------------------------------------- GEMM2: score2 = xn @ mn2^T
__global__ __launch_bounds__(256) void gemm_score2(const bf16_t* __restrict__ xn,
                                                   const bf16_t* __restrict__ mn2,
                                                   float* __restrict__ score2) {
  __shared__ __align__(16) bf16_t btile[2][16 * BSTR];
  int tid = threadIdx.x, wave = tid >> 5, lane = tid & 31;
  int row0 = blockIdx.x * 128 + wave * 16;
  int lr = lane & 15, half = lane >> 4;
  bool leader = (__builtin_amdgcn_readfirstlane((int)threadIdx.x) == 0);

  if (leader) tdm_load_2d(mn2, &btile[0][0], CC / 2, 16, CC / 2, true);

  v16bf afrag[CC / 32];
#pragma unroll
  for (int ks = 0; ks < CC / 32; ++ks) afrag[ks] = load_a_frag(xn, CC, row0, ks * 32, lane);

  for (int jt = 0; jt < KK / 16; ++jt) {
    int cur = jt & 1;
    if (leader) {
      if (jt + 1 < KK / 16) {
        tdm_load_2d(mn2 + (size_t)(jt + 1) * 16 * CC, &btile[cur ^ 1][0], CC / 2, 16, CC / 2, true);
        __builtin_amdgcn_s_wait_tensorcnt(1);
      } else {
        __builtin_amdgcn_s_wait_tensorcnt(0);
      }
    }
    __syncthreads();
    const bf16_t* bb = &btile[cur][0];
    v8f acc = {};
    v16bf bcur = load_b_frag_lds(bb, BSTR, 0, lane);
#pragma unroll
    for (int ks = 0; ks < CC / 32; ++ks) {
      v16bf bnext = bcur;
      if (ks + 1 < CC / 32) bnext = load_b_frag_lds(bb, BSTR, (ks + 1) * 32, lane);
      acc = __builtin_amdgcn_wmma_f32_16x16x32_bf16(false, afrag[ks], false, bcur,
                                                    (short)0, acc, false, false);
      bcur = bnext;
    }
    int col = jt * 16 + lr;
    int r0 = row0 + half * 8;
#pragma unroll
    for (int r = 0; r < 8; ++r) score2[(size_t)(r0 + r) * KK + col] = acc[r];
    __syncthreads();
  }
}

// ---------------------------------------------------------------- scatter (segment sum)
__global__ void scatter_accum(const float* __restrict__ x, const int* __restrict__ embed_ind,
                              float* __restrict__ embed_sum, float* __restrict__ counts) {
  int row = blockIdx.x * (blockDim.x >> 5) + (threadIdx.x >> 5);
  int lane = threadIdx.x & 31;
  if (row >= NN) return;
  int j = embed_ind[row];
  if (lane == 0) atomicAdd(&counts[j], 1.0f);
  const float* p = x + (size_t)row * CC;
  float* q = embed_sum + (size_t)j * CC;
#pragma unroll
  for (int i = lane; i < CC; i += 32) atomicAdd(&q[i], p[i]);
}

// ss[c] = sum_k embed_sum[k][c] / (counts[k] + 1e-6)
__global__ void column_sum_mean(const float* __restrict__ embed_sum,
                                const float* __restrict__ counts,
                                float* __restrict__ ss) {
  __shared__ float red[256];
  int c = blockIdx.x, t = threadIdx.x;
  float s = 0.0f;
  for (int k = t; k < KK; k += 256)
    s += embed_sum[(size_t)k * CC + c] / (counts[k] + 1e-6f);
  red[t] = s; __syncthreads();
  for (int off = 128; off; off >>= 1) { if (t < off) red[t] += red[t + off]; __syncthreads(); }
  if (t == 0) ss[c] = red[0];
}

// m_new = m*RATE + (mean + (mean - ss)*2/(K*(K-1)))*(1-RATE); plus bf16 transposed copy
__global__ void compute_mnew(const float* __restrict__ m, const float* __restrict__ embed_sum,
                             const float* __restrict__ counts, const float* __restrict__ ss,
                             float* __restrict__ m_new, bf16_t* __restrict__ mnewT) {
  int idx = blockIdx.x * blockDim.x + threadIdx.x;
  if (idx >= KK * CC) return;
  int k = idx / CC, c = idx - k * CC;
  float mean = embed_sum[idx] / (counts[k] + 1e-6f);
  const float coef = 2.0f / ((float)KK * (float)(KK - 1));
  float upd = mean + (mean - ss[c]) * coef;
  float v = m[idx] * RATE_F + upd * (1.0f - RATE_F);
  m_new[idx] = v;
  mnewT[(size_t)c * KK + k] = (bf16_t)v;
}

// ---------------------------------------------------------------- GEMM3: out0 = score2 @ m_new
// Block = 2 waves x 16-row strips (32 rows). A (32x1024) staged f32->bf16 in padded
// LDS once. B tiles (16 cols x 1024 k from m_new^T, 32KB) TDM double-buffered with
// 16B-per-1024B padding; ~130KB LDS total. A and B fragments register-pipelined.
#define ASTR (KK + 8)          // A LDS stride (elements)
#define B3ROW 2080             // padded B row footprint in bytes (2048 + 2*16)
static __device__ __forceinline__ v16bf load_a3(const bf16_t* a_lds, int rowbase,
                                                int ks, int lane) {
  int half = lane >> 4, lr = lane & 15;
  const bf16_t* p = a_lds + (rowbase + lr) * ASTR + ks * 32 + half * 8;
  return frag_from_chunks(*(const v8bf*)p, *(const v8bf*)(p + 16));
}
static __device__ __forceinline__ v16bf load_b3(const unsigned char* b_raw,
                                                int ks, int lane) {
  int half = lane >> 4, lr = lane & 15;
  int inner = (ks * 32 + half * 8) * 2;
  inner += (inner >> 10) << 4;                         // TDM pad: +16B per 1024B
  const bf16_t* p = (const bf16_t*)(b_raw + lr * B3ROW + inner);
  return frag_from_chunks(*(const v8bf*)p, *(const v8bf*)(p + 16));
}
__global__ __launch_bounds__(64) void gemm_out(const float* __restrict__ score2,
                                               const bf16_t* __restrict__ mnewT,
                                               float* __restrict__ out) {
  __shared__ __align__(16) bf16_t a_lds[32 * ASTR];
  __shared__ __align__(16) unsigned char b_raw[2][16 * B3ROW];
  int tid = threadIdx.x;
  int row0 = blockIdx.x * 32;
  bool leader = (__builtin_amdgcn_readfirstlane((int)threadIdx.x) == 0);

  if (leader) tdm_load_2d(mnewT, &b_raw[0][0], KK / 2, 16, KK / 2, true);

  for (int i = tid; i < 32 * KK; i += 64) {
    int r = i >> 10, c = i & (KK - 1);
    a_lds[r * ASTR + c] = (bf16_t)score2[(size_t)(row0 + r) * KK + c];
  }
  __syncthreads();

  int wave = tid >> 5, lane = tid & 31;
  int lr = lane & 15, half = lane >> 4;
  int sbase = wave * 16;
  for (int ct = 0; ct < CC / 16; ++ct) {
    int cur = ct & 1;
    if (leader) {
      if (ct + 1 < CC / 16) {
        tdm_load_2d(mnewT + (size_t)(ct + 1) * 16 * KK, &b_raw[cur ^ 1][0], KK / 2, 16, KK / 2, true);
        __builtin_amdgcn_s_wait_tensorcnt(1);
      } else {
        __builtin_amdgcn_s_wait_tensorcnt(0);
      }
    }
    __syncthreads();
    const unsigned char* bb = &b_raw[cur][0];
    v8f acc = {};
    v16bf acur = load_a3(a_lds, sbase, 0, lane);
    v16bf bcur = load_b3(bb, 0, lane);
#pragma unroll 8
    for (int ks = 0; ks < KK / 32; ++ks) {
      v16bf anext = acur, bnext = bcur;
      if (ks + 1 < KK / 32) {
        anext = load_a3(a_lds, sbase, ks + 1, lane);
        bnext = load_b3(bb, ks + 1, lane);
      }
      acc = __builtin_amdgcn_wmma_f32_16x16x32_bf16(false, acur, false, bcur,
                                                    (short)0, acc, false, false);
      acur = anext; bcur = bnext;
    }
    int col = ct * 16 + lr;
    int r0 = row0 + sbase + half * 8;
#pragma unroll
    for (int r = 0; r < 8; ++r) out[(size_t)(r0 + r) * CC + col] = acc[r];
    __syncthreads();
  }
}

// out *= ||x_row|| / ||out_row||
__global__ void rescale_rows(float* __restrict__ out, const float* __restrict__ tend1) {
  int row = blockIdx.x * (blockDim.x >> 5) + (threadIdx.x >> 5);
  int lane = threadIdx.x & 31;
  if (row >= NN) return;
  float* p = out + (size_t)row * CC;
  float ss = 0.0f;
#pragma unroll
  for (int i = lane; i < CC; i += 32) { float v = p[i]; ss += v * v; }
#pragma unroll
  for (int off = 16; off; off >>= 1) ss += __shfl_xor(ss, off, 32);
  float param = tend1[row] / sqrtf(ss);
#pragma unroll
  for (int i = lane; i < CC; i += 32) p[i] *= param;
}

// ---------------------------------------------------------------- launch
extern "C" void kernel_launch(void* const* d_in, const int* in_sizes, int n_in,
                              void* d_out, int out_size, void* d_ws, size_t ws_size,
                              hipStream_t stream) {
  (void)in_sizes; (void)n_in; (void)out_size; (void)ws_size;
  const float* x = (const float*)d_in[0];   // (N, C) f32
  const float* m = (const float*)d_in[1];   // (K, C) f32
  float* out0   = (float*)d_out;                       // (N, C) f32
  float* score2 = (float*)d_out + (size_t)NN * CC;     // (N, K) f32

  uintptr_t w = (uintptr_t)d_ws;
  auto take = [&](size_t bytes) -> void* {
    uintptr_t p = (w + 255) & ~(uintptr_t)255; w = p + bytes; return (void*)p;
  };
  bf16_t* xn        = (bf16_t*)take((size_t)NN * CC * 2);
  float*  tend1     = (float*) take((size_t)NN * 4);
  bf16_t* mn        = (bf16_t*)take((size_t)KK * CC * 2);
  int*    embed_ind = (int*)   take((size_t)NN * 4);
  float*  counts    = (float*) take((size_t)KK * 4);
  float*  embed_sum = (float*) take((size_t)KK * CC * 4);
  float*  ssv       = (float*) take((size_t)CC * 4);
  float*  m_new     = (float*) take((size_t)KK * CC * 4);
  bf16_t* mnewT     = (bf16_t*)take((size_t)CC * KK * 2);
  bf16_t* mn2       = (bf16_t*)take((size_t)KK * CC * 2);

  // 0) zero accumulators
  zero_f32<<<(KK * CC + 255) / 256, 256, 0, stream>>>(embed_sum, KK * CC);
  zero_f32<<<(KK + 255) / 256, 256, 0, stream>>>(counts, KK);

  // 1) normalize x -> xn (bf16) + ||x||; normalize m -> mn (bf16)
  row_normalize512<<<NN / 8, 256, 0, stream>>>(x, xn, tend1, NN);
  row_normalize512<<<KK / 8, 256, 0, stream>>>(m, mn, nullptr, KK);

  // 2) score GEMM + fused argmax (TDM-staged B)
  gemm_score_argmax<<<NN / 128, 256, 0, stream>>>(xn, mn, embed_ind);

  // 3) segment sums
  scatter_accum<<<NN / 8, 256, 0, stream>>>(x, embed_ind, embed_sum, counts);

  // 4) diversity + EMA update (+ transposed bf16 copy), renormalize
  column_sum_mean<<<CC, 256, 0, stream>>>(embed_sum, counts, ssv);
  compute_mnew<<<(KK * CC + 255) / 256, 256, 0, stream>>>(m, embed_sum, counts, ssv,
                                                          m_new, mnewT);
  row_normalize512<<<KK / 8, 256, 0, stream>>>(m_new, mn2, nullptr, KK);

  // 5) score2 = xn @ mn2^T (output #2, TDM-staged B)
  gemm_score2<<<NN / 128, 256, 0, stream>>>(xn, mn2, score2);

  // 6) out0 = score2 @ m_new (TDM-staged B), then row rescale (folds last matmul)
  gemm_out<<<NN / 32, 64, 0, stream>>>(score2, mnewT, out0);
  rescale_rows<<<NN / 8, 256, 0, stream>>>(out0, tend1);
}
